// GAT2_84954453115007
// MI455X (gfx1250) — matrix operand order repaired and checked by reading, not compiled, not run on hardware
//
#include <hip/hip_runtime.h>
#include <cfloat>

// ---------------------------------------------------------------------------
// 2-layer GAT (PyG GATConv semantics) for gfx1250 / MI455X.
//  - GEMMs via v_wmma_f32_16x16x32_f16 (f16 inputs, f32 accumulate)
//  - segment softmax over dst via 3 edge passes (atomic max / add / weighted add)
// ---------------------------------------------------------------------------

typedef __attribute__((ext_vector_type(16))) _Float16 v16h;
typedef __attribute__((ext_vector_type(8)))  _Float16 v8h;
typedef __attribute__((ext_vector_type(8)))  float    v8f;

#define NEG_SLOPE 0.2f

// ------------------------- WMMA GEMM: C[n,64] = A[n,K] * W[K,64] -----------
// 256 threads = 8 waves; each wave computes a 16x64 tile. W staged in LDS as
// f16, transposed to [n][k] with +8-half row padding for bank-conflict-free
// ds_load_b128 of B fragments. N rows must be multiple of 16 (100000 is).
__global__ void gat_wmma_gemm(const float* __restrict__ A,
                              const float* __restrict__ W,
                              float* __restrict__ C,
                              int nrows, int K) {
  extern __shared__ _Float16 sB[];            // [64][K+8]
  const int ldst = K + 8;                     // halves; row stride 16B-aligned
  const int tid  = threadIdx.x;

  // cooperative transpose + f32->f16 convert of W[K][64] into sB[n][k]
  for (int i = tid; i < 64 * K; i += 256) {
    int n = i / K;
    int k = i - n * K;
    sB[n * ldst + k] = (_Float16)W[k * 64 + n];
  }
  __syncthreads();

  const int wave     = tid >> 5;
  const int lane     = tid & 31;
  const int tile_row = blockIdx.x * 128 + wave * 16;
  if (tile_row >= nrows) return;              // wave-uniform: EXEC stays full

  const int mrow = tile_row + (lane & 15);
  const int kh   = (lane >> 4) * 8;           // A-fragment K sub-offset (0/8)
  const int ksB  = (lane >> 4) * 8;           // B-fragment K sub-offset (0/8)
  const int ncol = lane & 15;

  v8f acc[4] = {};

  for (int k0 = 0; k0 < K; k0 += 32) {
    // A fragment: lane holds row mrow, K = {k0+kh..+7, k0+kh+16..+23}
    const float* ap = A + (size_t)mrow * K + k0 + kh;
    float4 f0 = *(const float4*)(ap + 0);
    float4 f1 = *(const float4*)(ap + 4);
    float4 f2 = *(const float4*)(ap + 16);
    float4 f3 = *(const float4*)(ap + 20);
    v16h a;
    a[0]  = (_Float16)f0.x; a[1]  = (_Float16)f0.y;
    a[2]  = (_Float16)f0.z; a[3]  = (_Float16)f0.w;
    a[4]  = (_Float16)f1.x; a[5]  = (_Float16)f1.y;
    a[6]  = (_Float16)f1.z; a[7]  = (_Float16)f1.w;
    a[8]  = (_Float16)f2.x; a[9]  = (_Float16)f2.y;
    a[10] = (_Float16)f2.z; a[11] = (_Float16)f2.w;
    a[12] = (_Float16)f3.x; a[13] = (_Float16)f3.y;
    a[14] = (_Float16)f3.z; a[15] = (_Float16)f3.w;

#pragma unroll
    for (int nt = 0; nt < 4; ++nt) {
      // B fragment: lane holds column ncol(+nt*16), K = {k0+ksB..+7, k0+ksB+16..+23}
      const _Float16* bp = sB + (size_t)(nt * 16 + ncol) * ldst + k0 + ksB;
      v8h lo = *(const v8h*)(bp);
      v8h hi = *(const v8h*)(bp + 16);
      v16h b;
#pragma unroll
      for (int t = 0; t < 8; ++t) { b[t] = lo[t]; b[t + 8] = hi[t]; }
      acc[nt] = __builtin_amdgcn_wmma_f32_16x16x32_f16(
          false, a, false, b, (short)0, acc[nt], false, false);
    }
  }

  // C/D layout: VGPR r -> row tile_row + r + (lane>=16 ? 8 : 0), col = ncol
  const int rbase = tile_row + (lane >> 4) * 8;
#pragma unroll
  for (int nt = 0; nt < 4; ++nt)
#pragma unroll
    for (int r = 0; r < 8; ++r)
      C[(size_t)(rbase + r) * 64 + nt * 16 + ncol] = acc[nt][r];
}

// ------------------------- per-node attention scalars ----------------------
// a_s[n,h] = sum_c h[n,h,c]*att_src[h,c]; a_d analogous.  (H=4, C=16)
__global__ void gat_attn_scores(const float* __restrict__ h,
                                const float* __restrict__ att_src,
                                const float* __restrict__ att_dst,
                                float* __restrict__ a_s,
                                float* __restrict__ a_d, int n) {
  int i = blockIdx.x * blockDim.x + threadIdx.x;
  if (i >= n) return;
  const float4* hv = (const float4*)(h + (size_t)i * 64);
#pragma unroll
  for (int hd = 0; hd < 4; ++hd) {
    float ss = 0.f, dd = 0.f;
#pragma unroll
    for (int q = 0; q < 4; ++q) {
      float4 v = hv[hd * 4 + q];
      const float* as = att_src + hd * 16 + q * 4;
      const float* ad = att_dst + hd * 16 + q * 4;
      ss += v.x * as[0] + v.y * as[1] + v.z * as[2] + v.w * as[3];
      dd += v.x * ad[0] + v.y * ad[1] + v.z * ad[2] + v.w * ad[3];
    }
    a_s[i * 4 + hd] = ss;
    a_d[i * 4 + hd] = dd;
  }
}

// ------------------------- fills ------------------------------------------
__global__ void gat_fill(float* __restrict__ p, float v, int n) {
  int i = blockIdx.x * blockDim.x + threadIdx.x;
  if (i < n) p[i] = v;
}

// ------------------------- float atomic max (bit trick) --------------------
__device__ __forceinline__ void atomicMaxF(float* addr, float v) {
  if (v >= 0.0f) atomicMax((int*)addr, __float_as_int(v));
  else           atomicMin((unsigned int*)addr, __float_as_uint(v));
}

__device__ __forceinline__ float lrelu(float v) {
  return v > 0.0f ? v : NEG_SLOPE * v;
}

// ------------------------- edge pass 1: segment max ------------------------
__global__ void gat_edge_max(const int* __restrict__ esrc,
                             const int* __restrict__ edst, int E, int Etot,
                             const float* __restrict__ a_s,
                             const float* __restrict__ a_d,
                             float* __restrict__ m) {
  int i = blockIdx.x * blockDim.x + threadIdx.x;
  if (i >= Etot) return;
  int s = (i < E) ? esrc[i] : (i - E);     // self loops appended
  int d = (i < E) ? edst[i] : (i - E);
#pragma unroll
  for (int hd = 0; hd < 4; ++hd) {
    float e = lrelu(a_s[s * 4 + hd] + a_d[d * 4 + hd]);
    atomicMaxF(&m[d * 4 + hd], e);
  }
}

// ------------------------- edge pass 2: segment sum of exp -----------------
__global__ void gat_edge_sum(const int* __restrict__ esrc,
                             const int* __restrict__ edst, int E, int Etot,
                             const float* __restrict__ a_s,
                             const float* __restrict__ a_d,
                             const float* __restrict__ m,
                             float* __restrict__ ssum) {
  int i = blockIdx.x * blockDim.x + threadIdx.x;
  if (i >= Etot) return;
  int s = (i < E) ? esrc[i] : (i - E);
  int d = (i < E) ? edst[i] : (i - E);
#pragma unroll
  for (int hd = 0; hd < 4; ++hd) {
    float e = lrelu(a_s[s * 4 + hd] + a_d[d * 4 + hd]);
    atomicAdd(&ssum[d * 4 + hd], __expf(e - m[d * 4 + hd]));
  }
}

// ------------------------- edge pass 3: weighted aggregation ---------------
// 16 lanes per edge; lane j moves float4 #j of the 64-float message.
__global__ void gat_edge_agg(const int* __restrict__ esrc,
                             const int* __restrict__ edst, int E, int Etot,
                             const float* __restrict__ a_s,
                             const float* __restrict__ a_d,
                             const float* __restrict__ m,
                             const float* __restrict__ ssum,
                             const float* __restrict__ hin,
                             float* __restrict__ out) {
  int gid = blockIdx.x * blockDim.x + threadIdx.x;
  int e = gid >> 4;
  int j = gid & 15;
  if (e >= Etot) return;
  int s = (e < E) ? esrc[e] : (e - E);
  int d = (e < E) ? edst[e] : (e - E);
  int hd = j >> 2;                               // 16 channels per head
  float ev = lrelu(a_s[s * 4 + hd] + a_d[d * 4 + hd]);
  float alpha = __expf(ev - m[d * 4 + hd]) / (ssum[d * 4 + hd] + 1e-16f);
  float4 hv = ((const float4*)(hin + (size_t)s * 64))[j];
  float* op = out + (size_t)d * 64 + j * 4;
  atomicAdd(op + 0, hv.x * alpha);
  atomicAdd(op + 1, hv.y * alpha);
  atomicAdd(op + 2, hv.z * alpha);
  atomicAdd(op + 3, hv.w * alpha);
}

// ------------------------- bias (+optional relu) ---------------------------
__global__ void gat_bias_act(float* __restrict__ h, const float* __restrict__ b,
                             int n, int do_relu) {
  int i = blockIdx.x * blockDim.x + threadIdx.x;
  if (i >= n * 64) return;
  float v = h[i] + b[i & 63];
  if (do_relu) v = v > 0.0f ? v : 0.0f;
  h[i] = v;
}

// ---------------------------------------------------------------------------
extern "C" void kernel_launch(void* const* d_in, const int* in_sizes, int n_in,
                              void* d_out, int out_size, void* d_ws, size_t ws_size,
                              hipStream_t stream) {
  const float* x   = (const float*)d_in[0];
  const int*   eix = (const int*)  d_in[1];
  const float* W1  = (const float*)d_in[2];
  const float* as1 = (const float*)d_in[3];
  const float* ad1 = (const float*)d_in[4];
  const float* b1  = (const float*)d_in[5];
  const float* W2  = (const float*)d_in[6];
  const float* as2 = (const float*)d_in[7];
  const float* ad2 = (const float*)d_in[8];
  const float* b2  = (const float*)d_in[9];
  float* out = (float*)d_out;

  const int N    = in_sizes[0] / 256;      // IN = 256
  const int E    = in_sizes[1] / 2;
  const int Etot = E + N;                  // + self loops
  const int* esrc = eix;
  const int* edst = eix + E;

  // workspace layout (floats)
  float* ws   = (float*)d_ws;
  float* h    = ws;                         // [N,64] GEMM output (both layers)
  float* hr   = h  + (size_t)N * 64;        // [N,64] layer-1 aggregated/relu
  float* a_s  = hr + (size_t)N * 64;        // [N,4]
  float* a_d  = a_s + (size_t)N * 4;        // [N,4]
  float* mbuf = a_d + (size_t)N * 4;        // [N,4]
  float* sbuf = mbuf + (size_t)N * 4;       // [N,4]

  const int gemm_grid  = (N + 127) / 128;
  const int node_grid  = (N + 255) / 256;
  const int nh_grid    = (N * 4 + 255) / 256;
  const int feat_grid  = (N * 64 + 255) / 256;
  const int edge_grid  = (Etot + 255) / 256;
  const int agg_grid   = (Etot * 16 + 255) / 256;

  // ======================== Layer 1 ========================
  gat_wmma_gemm<<<gemm_grid, 256, 64 * (256 + 8) * sizeof(_Float16), stream>>>(
      x, W1, h, N, 256);
  gat_attn_scores<<<node_grid, 256, 0, stream>>>(h, as1, ad1, a_s, a_d, N);

  gat_fill<<<nh_grid,   256, 0, stream>>>(mbuf, -FLT_MAX, N * 4);
  gat_fill<<<nh_grid,   256, 0, stream>>>(sbuf, 0.0f,     N * 4);
  gat_fill<<<feat_grid, 256, 0, stream>>>(hr,   0.0f,     N * 64);

  gat_edge_max<<<edge_grid, 256, 0, stream>>>(esrc, edst, E, Etot, a_s, a_d, mbuf);
  gat_edge_sum<<<edge_grid, 256, 0, stream>>>(esrc, edst, E, Etot, a_s, a_d, mbuf, sbuf);
  gat_edge_agg<<<agg_grid,  256, 0, stream>>>(esrc, edst, E, Etot, a_s, a_d,
                                              mbuf, sbuf, h, hr);
  gat_bias_act<<<feat_grid, 256, 0, stream>>>(hr, b1, N, 1);

  // ======================== Layer 2 ========================
  gat_wmma_gemm<<<gemm_grid, 256, 64 * (64 + 8) * sizeof(_Float16), stream>>>(
      hr, W2, h, N, 64);
  gat_attn_scores<<<node_grid, 256, 0, stream>>>(h, as2, ad2, a_s, a_d, N);

  gat_fill<<<nh_grid,   256, 0, stream>>>(mbuf, -FLT_MAX, N * 4);
  gat_fill<<<nh_grid,   256, 0, stream>>>(sbuf, 0.0f,     N * 4);
  gat_fill<<<feat_grid, 256, 0, stream>>>(out,  0.0f,     N * 64);

  gat_edge_max<<<edge_grid, 256, 0, stream>>>(esrc, edst, E, Etot, a_s, a_d, mbuf);
  gat_edge_sum<<<edge_grid, 256, 0, stream>>>(esrc, edst, E, Etot, a_s, a_d, mbuf, sbuf);
  gat_edge_agg<<<agg_grid,  256, 0, stream>>>(esrc, edst, E, Etot, a_s, a_d,
                                              mbuf, sbuf, h, out);
  gat_bias_act<<<feat_grid, 256, 0, stream>>>(out, b2, N, 0);
}
